// TimeCrossAttn_68685116998214
// MI455X (gfx1250) — matrix operand order repaired and checked by reading, not compiled, not run on hardware
//
#include <hip/hip_runtime.h>
#include <hip/hip_bf16.h>

// ---------------------------------------------------------------------------
// TimeCrossAttn, algebraically collapsed (q has no spatial dim):
//   qk[b,h,:]   = (q_bh^T Wk) / sqrt(d)            (tiny)
//   logits      = qk . x  + qbk                     (WMMA bf16, 0.4 GF)
//   attn        = softmax_m(logits)
//   y[b,h,:]    = x . attn^T                        (WMMA bf16, 0.4 GF)
//   out768      = Wv . y + bv   ;  final = Wo . out768 + bo   (tiny)
//   out[b,c,:,:]= cond + final[b,c]                 (bandwidth bound)
// ---------------------------------------------------------------------------

typedef __attribute__((ext_vector_type(16))) __bf16 v16bf;
typedef __attribute__((ext_vector_type(8)))  __bf16 v8bf;
typedef __attribute__((ext_vector_type(8)))  float  v8f;

#define B_   16
#define C_   512
#define HW_  4096
#define T_   768
#define NH_  12

__device__ __forceinline__ __bf16 tobf(float f) { return (__bf16)f; }

// ---- Kernel A: qk[b,16,512] (bf16, rows 12..15 zero), qbk16[b,16] (zero-pad)
__global__ __launch_bounds__(256) void qk_proj_kernel(
    const float* __restrict__ t, const float* __restrict__ Wk,
    const float* __restrict__ bk,
    unsigned short* __restrict__ qkbf_raw, float* __restrict__ qbk16)
{
    __bf16* qkbf = (__bf16*)qkbf_raw;
    int idx = blockIdx.x * 256 + threadIdx.x;       // B*16*512 = 131072
    int c   = idx & (C_ - 1);
    int hp  = (idx >> 9) & 15;
    int b   = idx >> 13;
    if (hp >= NH_) {
        qkbf[idx] = tobf(0.0f);
        if (c == 0) qbk16[b * 16 + hp] = 0.0f;      // padded bias rows = 0
        return;
    }
    const float* tb = t + b * T_ + hp * 64;
    float s = 0.0f;
    #pragma unroll 8
    for (int d = 0; d < 64; ++d)
        s += tb[d] * Wk[(size_t)(hp * 64 + d) * C_ + c];
    qkbf[idx] = tobf(s * 0.125f);                   // fold 1/sqrt(64)
    if (c == 0) {
        float sb = 0.0f;
        for (int d = 0; d < 64; ++d) sb += tb[d] * bk[hp * 64 + d];
        qbk16[b * 16 + hp] = sb * 0.125f;
    }
}

// ---- Kernel B: logits[b,16,4096] = qk(16x512) x x(512x4096) via WMMA --------
__global__ __launch_bounds__(256) void logits_kernel(
    const float* __restrict__ x, const unsigned short* __restrict__ qkbf_raw,
    const float* __restrict__ qbk16, float* __restrict__ attn)
{
    const __bf16* qkbf = (const __bf16*)qkbf_raw;
    const int b    = blockIdx.y;
    const int wave = threadIdx.x >> 5;
    const int lane = threadIdx.x & 31;
    const int m0   = (blockIdx.x * 8 + wave) * 16;  // 256 m-tiles / batch
    const int half = lane >> 4;
    const int l16  = lane & 15;
    const int ab   = half * 8;                      // A K-base per lane half
    const int kb   = half * 16;                     // unused: B's K-base is ab, same as A

    const __bf16* arow = qkbf + ((size_t)b * 16 + l16) * C_;
    const float*  xb   = x + (size_t)b * C_ * HW_;
    const int     col  = m0 + l16;

    v8f acc = {};
    #pragma unroll 2
    for (int k0 = 0; k0 < C_; k0 += 32) {
        v16bf a;
        {   // two contiguous 8-element runs (ISA 16-bit A layout)
            v8bf r1 = *(const v8bf*)(arow + k0 + ab);
            v8bf r2 = *(const v8bf*)(arow + k0 + 16 + ab);
            #pragma unroll
            for (int j = 0; j < 8; ++j) { a[j] = r1[j]; a[8 + j] = r2[j]; }
        }
        v16bf bf;   // B: column = col, K strided down rows of x
        #pragma unroll
        for (int j = 0; j < 16; ++j)
            bf[j] = tobf(xb[(size_t)(k0 + ((j & 8) << 1) + ab + (j & 7)) * HW_ + col]);
        acc = __builtin_amdgcn_wmma_f32_16x16x32_bf16(
                  false, a, false, bf, (short)0, acc, false, false);
    }
    // uniform bias epilogue: lane's 8 rows M = r + half*8 are contiguous
    const v8f qb = *(const v8f*)(qbk16 + b * 16 + half * 8);
    #pragma unroll
    for (int r = 0; r < 8; ++r) {
        int M = r + half * 8;
        attn[((size_t)b * 16 + M) * HW_ + col] = acc[r] + qb[r];
    }
}

// ---- Kernel C: row softmax over 4096, one block per real (b,h) row ----------
__global__ __launch_bounds__(256) void softmax_kernel(float* __restrict__ attn)
{
    int b = blockIdx.x / NH_, h = blockIdx.x % NH_;
    float* row = attn + ((size_t)b * 16 + h) * HW_;
    int tid = threadIdx.x;
    float vals[16];
    float m = -1e30f;
    #pragma unroll
    for (int i = 0; i < 16; ++i) { vals[i] = row[tid + i * 256]; m = fmaxf(m, vals[i]); }
    __shared__ float red[256];
    red[tid] = m; __syncthreads();
    for (int s = 128; s > 0; s >>= 1) {
        if (tid < s) red[tid] = fmaxf(red[tid], red[tid + s]);
        __syncthreads();
    }
    m = red[0]; __syncthreads();
    float sum = 0.0f;
    #pragma unroll
    for (int i = 0; i < 16; ++i) { vals[i] = __expf(vals[i] - m); sum += vals[i]; }
    red[tid] = sum; __syncthreads();
    for (int s = 128; s > 0; s >>= 1) {
        if (tid < s) red[tid] += red[tid + s];
        __syncthreads();
    }
    float inv = 1.0f / red[0];
    #pragma unroll
    for (int i = 0; i < 16; ++i) row[tid + i * 256] = vals[i] * inv;
}

// ---- Kernel D: y[b,16,512] = attn(16x4096) x x^T(4096x512) via WMMA ---------
__global__ __launch_bounds__(256) void av_kernel(
    const float* __restrict__ x, const float* __restrict__ attn,
    float* __restrict__ y)
{
    const int b    = blockIdx.y;
    const int wave = threadIdx.x >> 5;
    const int lane = threadIdx.x & 31;
    const int c0   = (blockIdx.x * 8 + wave) * 16;  // 32 c-tiles / batch
    const int half = lane >> 4;
    const int l16  = lane & 15;
    const int ab   = half * 8;
    const int kb   = half * 16;

    const float* arow = attn + ((size_t)b * 16 + l16) * HW_;
    const float* xrow = x + ((size_t)b * C_ + c0 + l16) * HW_;

    v8f acc = {};
    #pragma unroll 2
    for (int k0 = 0; k0 < HW_; k0 += 32) {
        v16bf a;
        #pragma unroll
        for (int j = 0; j < 8; ++j) {
            a[j]     = tobf(arow[k0 + ab + j]);
            a[8 + j] = tobf(arow[k0 + 16 + ab + j]);
        }
        v16bf bf;   // B[K=m][N=c]: contiguous along m within row c0+l16
        #pragma unroll
        for (int j = 0; j < 16; ++j)
            bf[j] = tobf(xrow[k0 + ((j & 8) << 1) + ab + (j & 7)]);
        acc = __builtin_amdgcn_wmma_f32_16x16x32_bf16(
                  false, a, false, bf, (short)0, acc, false, false);
    }
    #pragma unroll
    for (int r = 0; r < 8; ++r) {
        int M = r + half * 8;
        y[((size_t)b * 16 + M) * C_ + c0 + l16] = acc[r];
    }
}

// ---- Kernel E1: out768[b,t] = Wv[t,:] . y[b,h(t),:] + bv[t] -----------------
__global__ __launch_bounds__(256) void vproj_kernel(
    const float* __restrict__ Wv, const float* __restrict__ bv,
    const float* __restrict__ y, float* __restrict__ out768)
{
    int idx = blockIdx.x * 256 + threadIdx.x;       // 16*768
    int tt  = idx % T_;
    int b   = idx / T_;
    int h   = tt >> 6;
    const float* yrow = y + ((size_t)b * 16 + h) * C_;
    const float* wrow = Wv + (size_t)tt * C_;
    float s = bv[tt];
    #pragma unroll 8
    for (int c = 0; c < C_; ++c) s += wrow[c] * yrow[c];
    out768[idx] = s;
}

// ---- Kernel E2: final[b,c] = Wo[c,:] . out768[b,:] + bo[c] ------------------
__global__ __launch_bounds__(256) void oproj_kernel(
    const float* __restrict__ Wo, const float* __restrict__ bo,
    const float* __restrict__ out768, float* __restrict__ fin)
{
    int idx = blockIdx.x * 256 + threadIdx.x;       // 16*512
    int c   = idx & (C_ - 1);
    int b   = idx >> 9;
    const float* wrow = Wo + (size_t)c * T_;
    const float* ob   = out768 + b * T_;
    float s = bo[c];
    #pragma unroll 8
    for (int tt = 0; tt < T_; ++tt) s += wrow[tt] * ob[tt];
    fin[idx] = s;
}

// ---- Kernel F: out[b,c,:,:] = cond + final[b,c]  (float4, bandwidth bound) --
__global__ __launch_bounds__(256) void add_kernel(
    const float* __restrict__ cond, const float* __restrict__ fin,
    float* __restrict__ out)
{
    size_t i4  = (size_t)blockIdx.x * 256 + threadIdx.x;   // 8388608 float4
    size_t row = i4 >> 10;                                 // 1024 f4 per (b,c)
    float  add = fin[row];
    float4 v   = ((const float4*)cond)[i4];
    float4 o   = make_float4(v.x + add, v.y + add, v.z + add, v.w + add);
    ((float4*)out)[i4] = o;
}

extern "C" void kernel_launch(void* const* d_in, const int* in_sizes, int n_in,
                              void* d_out, int out_size, void* d_ws, size_t ws_size,
                              hipStream_t stream) {
    const float* cond = (const float*)d_in[0];
    const float* t    = (const float*)d_in[1];
    const float* Wk   = (const float*)d_in[2];
    const float* bk   = (const float*)d_in[3];
    const float* Wv   = (const float*)d_in[4];
    const float* bv   = (const float*)d_in[5];
    const float* Wo   = (const float*)d_in[6];
    const float* bo   = (const float*)d_in[7];
    float* out = (float*)d_out;

    char* ws = (char*)d_ws;
    unsigned short* qkbf   = (unsigned short*)(ws + 0);            // 256 KB
    float*          qbk16  = (float*)(ws + 262144);                //   1 KB (B x 16, zero-padded)
    float*          attn   = (float*)(ws + 263168);                //   4 MB
    float*          y      = (float*)(ws + 4457472);               // 512 KB
    float*          out768 = (float*)(ws + 4981760);               //  48 KB
    float*          fin    = (float*)(ws + 5030912);               //  32 KB

    qk_proj_kernel<<<dim3(512), dim3(256), 0, stream>>>(t, Wk, bk, qkbf, qbk16);
    logits_kernel <<<dim3(32, B_), dim3(256), 0, stream>>>(cond, qkbf, qbk16, attn);
    softmax_kernel<<<dim3(B_ * NH_), dim3(256), 0, stream>>>(attn);
    av_kernel     <<<dim3(4, B_), dim3(256), 0, stream>>>(cond, attn, y);
    vproj_kernel  <<<dim3(48), dim3(256), 0, stream>>>(Wv, bv, y, out768);
    oproj_kernel  <<<dim3(32), dim3(256), 0, stream>>>(Wo, bo, out768, fin);
    add_kernel    <<<dim3(32768), dim3(256), 0, stream>>>(cond, fin, out);
}
